// GIN_50929722196056
// MI455X (gfx1250) — matrix operand order repaired and checked by reading, not compiled, not run on hardware
//
#include <hip/hip_runtime.h>
#include <hip/hip_bf16.h>

#define N_NODES 100000
#define N_EDGES 800000

typedef __attribute__((ext_vector_type(16))) _Float16 v16h;
typedef __attribute__((ext_vector_type(8)))  float    v8f;

union Frag16 { v16h h; uint32_t u[8]; uint4 q[2]; };

__device__ __forceinline__ v8f wmma_f16f32(v16h a, v16h b, v8f c) {
  // D(f32 16x16) = A(f16 16x32) * B(f16 32x16) + C
  return __builtin_amdgcn_wmma_f32_16x16x32_f16(false, a, false, b, (short)0, c,
                                                false, false);
}

// B-matrix fragment: 16 contiguous halves (column of W, K-slice per lane half)
__device__ __forceinline__ v16h load_b_frag(const _Float16* p) {
  Frag16 f;
  f.q[0] = *(const uint4*)(p);
  f.q[1] = *(const uint4*)(p + 8);
  return f.h;
}
// A-matrix fragment: row base + two 8-half chunks at kb and kb+16
__device__ __forceinline__ v16h load_a_frag(const _Float16* row, int kb) {
  Frag16 f;
  f.q[0] = *(const uint4*)(row + kb);
  f.q[1] = *(const uint4*)(row + kb + 16);
  return f.h;
}

// Branch-free TRANS-pipe activations (co-issue with WMMA on CDNA5)
__device__ __forceinline__ float fast_exp2(float x) {
#if __has_builtin(__builtin_amdgcn_exp2f)
  return __builtin_amdgcn_exp2f(x);
#else
  return exp2f(x);
#endif
}
__device__ __forceinline__ float fast_rcp(float x) {
#if __has_builtin(__builtin_amdgcn_rcpf)
  return __builtin_amdgcn_rcpf(x);
#else
  return 1.0f / x;
#endif
}
__device__ __forceinline__ float fast_sigmoid(float x) {
  return fast_rcp(1.0f + fast_exp2(-1.44269504f * x));
}
__device__ __forceinline__ float fast_tanh(float x) {
#if __has_builtin(__builtin_amdgcn_tanhf)
  return __builtin_amdgcn_tanhf(x);
#else
  return fmaf(2.0f, fast_sigmoid(2.0f * x), -1.0f);
#endif
}

// ---------------------------------------------------------------------------
// init: zero agg + out, convert message-MLP weights to f16, and pre-pack the
// LSTM weights:   whh16 = f16 Whh [128][32] (A-fragment friendly)
//                 wihp  = f16 [128][8]: cols 0..5 = Wih, col 6 = b_ih+b_hh
// ---------------------------------------------------------------------------
__global__ void __launch_bounds__(256) init_kernel(
    float* __restrict__ agg, float* __restrict__ out,
    const float* __restrict__ w1, const float* __restrict__ w2,
    _Float16* __restrict__ w1h, _Float16* __restrict__ w2h,
    const float* __restrict__ whh_f, const float* __restrict__ whh_b,
    const float* __restrict__ wih_f, const float* __restrict__ wih_b,
    const float* __restrict__ bih_f, const float* __restrict__ bhh_f,
    const float* __restrict__ bih_b, const float* __restrict__ bhh_b,
    _Float16* __restrict__ whh16f, _Float16* __restrict__ whh16b,
    _Float16* __restrict__ wihpf, _Float16* __restrict__ wihpb) {
  const long i = (long)blockIdx.x * blockDim.x + threadIdx.x;
  const long tot = (long)N_NODES * 32;
  for (long k = i; k < tot; k += (long)gridDim.x * blockDim.x) agg[k] = 0.0f;
  if (i < 64) out[i] = 0.0f;
  if (i < 32 * 96) w1h[i] = (_Float16)w1[i];
  if (i < 32 * 32) w2h[i] = (_Float16)w2[i];
  if (i < 128 * 32) {
    whh16f[i] = (_Float16)whh_f[i];
    whh16b[i] = (_Float16)whh_b[i];
  }
  if (i < 128 * 8) {
    const int g = (int)(i >> 3), k = (int)(i & 7);
    float vf = 0.0f, vb = 0.0f;
    if (k < 6)      { vf = wih_f[g * 6 + k];      vb = wih_b[g * 6 + k]; }
    else if (k == 6){ vf = bih_f[g] + bhh_f[g];   vb = bih_b[g] + bhh_b[g]; }
    wihpf[i] = (_Float16)vf;
    wihpb[i] = (_Float16)vb;
  }
}

// ---------------------------------------------------------------------------
// coor = MLP(x): [N,3] -> relu -> [N,32] -> [N,32], stored f16
// ---------------------------------------------------------------------------
__global__ void __launch_bounds__(256) coor_kernel(
    const float* __restrict__ x, const float* __restrict__ w1,
    const float* __restrict__ w2, _Float16* __restrict__ coor16) {
  const long n = (long)blockIdx.x * blockDim.x + threadIdx.x;
  if (n >= N_NODES) return;
  const float x0 = x[n * 3 + 0], x1 = x[n * 3 + 1], x2 = x[n * 3 + 2];
  float hid[32];
#pragma unroll
  for (int o = 0; o < 32; ++o)
    hid[o] = fmaxf(fmaf(w1[o * 3 + 2], x2, fmaf(w1[o * 3 + 1], x1, w1[o * 3] * x0)), 0.0f);
#pragma unroll 4
  for (int o = 0; o < 32; ++o) {
    float s = 0.0f;
#pragma unroll
    for (int k = 0; k < 32; ++k) s = fmaf(hid[k], w2[o * 32 + k], s);
    coor16[n * 32 + o] = (_Float16)s;
  }
}

// ---------------------------------------------------------------------------
// Bidirectional LSTM over edge trajectories.
// One wave handles 16 edges. Per step:  G(128x16) = Whh x h  +  Wihp x [x;1]
// via 16 v_wmma_f32_16x16x32_f16 (K=32 == H exactly, bias folded in col 6).
// D layout: lane L<16 owns gates {16t+r} of edge L at hidden {0..7,16..23};
// lane L+16 owns hidden {8..15,24..31} of edge L -> gate update is lane-local,
// next-step B(h) fragment is the lane's own h, no swap needed.
// ---------------------------------------------------------------------------
__global__ void __launch_bounds__(256) lstm_kernel(
    const float* __restrict__ traj,
    const _Float16* __restrict__ whh16f, const _Float16* __restrict__ whh16b,
    const _Float16* __restrict__ wihpf,  const _Float16* __restrict__ wihpb,
    _Float16* __restrict__ traj16) {
  const int lane = threadIdx.x & 31;
  const int wid  = blockIdx.x * (blockDim.x >> 5) + (threadIdx.x >> 5);
  const int el   = lane & 15;
  const bool hi  = lane >= 16;
  const int kb   = hi ? 8 : 0;
  const long e   = (long)wid * 16 + el;
  const uint4 z4 = make_uint4(0u, 0u, 0u, 0u);

  float cf[16], res[16];

  for (int d = 0; d < 2; ++d) {
    const _Float16* whh16 = d ? whh16b : whh16f;
    const _Float16* wihp  = d ? wihpb : wihpf;

    v16h wa[8], xa[8];   // A fragments: Whh and [Wih|bias] per 16-gate tile
#pragma unroll
    for (int t = 0; t < 8; ++t) {
      const int g = t * 16 + el;  // A row = gate
      wa[t] = load_a_frag(whh16 + g * 32, kb);
      Frag16 fx;
      fx.q[0] = hi ? z4 : *(const uint4*)(wihp + g * 8);  // K 0..7 (6 feats+bias)
      fx.q[1] = z4;                                       // K 16..23 / hi lanes: 0
      xa[t] = fx.h;
    }

    float c_own[16], h_own[16];
#pragma unroll
    for (int j = 0; j < 16; ++j) { c_own[j] = 0.0f; h_own[j] = 0.0f; }
    Frag16 bh;
#pragma unroll
    for (int p = 0; p < 8; ++p) bh.u[p] = 0u;

    for (int s = 0; s < 8; ++s) {
      const int t = d ? (7 - s) : s;
      // B fragment of [x_t;1]: features in K 0..5, constant 1 in K 6 (lo lanes)
      Frag16 bx;
#pragma unroll
      for (int p = 0; p < 8; ++p) bx.u[p] = 0u;
      if (!hi) {
        const float* xp = traj + e * 48 + t * 6;
#pragma unroll
        for (int j = 0; j < 6; ++j) bx.h[j] = (_Float16)xp[j];
        bx.h[6] = (_Float16)1.0f;
        if (s < 7) {
          const int tn = d ? (6 - s) : (s + 1);
          __builtin_prefetch(traj + e * 48 + tn * 6, 0, 1);
        }
      }
      v8f C[8];
#pragma unroll
      for (int nt = 0; nt < 8; ++nt) {
        v8f z;
#pragma unroll
        for (int r = 0; r < 8; ++r) z[r] = 0.0f;
        C[nt] = wmma_f16f32(xa[nt], bx.h, z);      // x @ Wih^T + bias
        C[nt] = wmma_f16f32(wa[nt], bh.h, C[nt]);  // + h @ Whh^T
      }
      // lane-local gate nonlinearity + state update (pytorch order i,f,g,o)
#pragma unroll
      for (int j = 0; j < 16; ++j) {
        const int tl = (j < 8) ? 0 : 1;
        const int r  = j & 7;
        const float gi = fast_sigmoid(C[0 + tl][r]);
        const float gf = fast_sigmoid(C[2 + tl][r]);
        const float gg = fast_tanh(C[4 + tl][r]);
        const float go = fast_sigmoid(C[6 + tl][r]);
        c_own[j] = gf * c_own[j] + gi * gg;
        h_own[j] = go * fast_tanh(c_own[j]);
      }
      // rebuild B(h) fragment lane-locally: own h already in B's k order
      Frag16 own;
#pragma unroll
      for (int j = 0; j < 16; ++j) own.h[j] = (_Float16)h_own[j];
      uint32_t ex[8];
#pragma unroll
      for (int p = 0; p < 8; ++p)
        ex[p] = (uint32_t)__shfl_xor((int)own.u[p], 16, 32);
#pragma unroll
      for (int p = 0; p < 4; ++p) {
        bh.u[p]     = own.u[p];      // K 0..7  / 8..15
        bh.u[4 + p] = own.u[4 + p];  // K 16..23 / 24..31
      }
    }
    if (d == 0) {
#pragma unroll
      for (int j = 0; j < 16; ++j) cf[j] = c_own[j];
    } else {
#pragma unroll
      for (int j = 0; j < 16; ++j) res[j] = 0.5f * (cf[j] + c_own[j]);
    }
  }
  // store traj encoding f16 [E][32]: two 16B chunks per lane
  Frag16 o;
#pragma unroll
  for (int j = 0; j < 16; ++j) o.h[j] = (_Float16)res[j];
  _Float16* outp = traj16 + e * 32 + kb;
  *(uint4*)(outp)      = o.q[0];   // hidden kb..kb+7
  *(uint4*)(outp + 16) = o.q[1];   // hidden kb+16..kb+23
}

// ---------------------------------------------------------------------------
// Message MLP: msg = W2 relu(W1 [coor[dst]|coor[src]|traj]) , scatter into agg
// A = per-edge rows (16 edges/wave), B = weight columns. K-blocks of 32.
// ---------------------------------------------------------------------------
__global__ void __launch_bounds__(256) msg_kernel(
    const int* __restrict__ eidx, const _Float16* __restrict__ coor16,
    const _Float16* __restrict__ traj16, const _Float16* __restrict__ w1h,
    const _Float16* __restrict__ w2h, float* __restrict__ agg) {
  __shared__ _Float16 lds[8][16 * 32];
  const int lane = threadIdx.x & 31;
  const int wv   = threadIdx.x >> 5;
  const int wid  = blockIdx.x * 8 + wv;
  const int el   = lane & 15;
  const bool hi  = lane >= 16;
  const int kb   = hi ? 8 : 0;
  const long e0  = (long)wid * 16;
  const long e   = e0 + el;

  v16h bw1[2][3], bw2[2];
#pragma unroll
  for (int nt = 0; nt < 2; ++nt) {
    const int col = nt * 16 + el;
#pragma unroll
    for (int kblk = 0; kblk < 3; ++kblk)
      bw1[nt][kblk] = load_a_frag(w1h + col * 96 + kblk * 32, kb);
    bw2[nt] = load_a_frag(w2h + col * 32, kb);
  }

  const int srcE = eidx[e];
  const int dstE = eidx[N_EDGES + e];
  v16h a[3];
  a[0] = load_a_frag(coor16 + (long)dstE * 32, kb);  // x_i
  a[1] = load_a_frag(coor16 + (long)srcE * 32, kb);  // x_j
  a[2] = load_a_frag(traj16 + e * 32, kb);           // edge_attr

  v8f h1[2];
#pragma unroll
  for (int nt = 0; nt < 2; ++nt) {
    v8f c;
#pragma unroll
    for (int r = 0; r < 8; ++r) c[r] = 0.0f;
#pragma unroll
    for (int kblk = 0; kblk < 3; ++kblk) c = wmma_f16f32(a[kblk], bw1[nt][kblk], c);
#pragma unroll
    for (int r = 0; r < 8; ++r) c[r] = fmaxf(c[r], 0.0f);
    h1[nt] = c;
  }
  // D layout -> A layout via per-wave LDS bounce
#pragma unroll
  for (int nt = 0; nt < 2; ++nt) {
    const int col = nt * 16 + el;
#pragma unroll
    for (int r = 0; r < 8; ++r) {
      const int m = r + (hi ? 8 : 0);
      lds[wv][m * 32 + col] = (_Float16)h1[nt][r];
    }
  }
  asm volatile("s_wait_dscnt 0x0" ::: "memory");  // same-wave LDS RAW fence
  v16h a2 = load_a_frag(&lds[wv][el * 32], kb);

#pragma unroll
  for (int nt = 0; nt < 2; ++nt) {
    v8f c;
#pragma unroll
    for (int r = 0; r < 8; ++r) c[r] = 0.0f;
    c = wmma_f16f32(a2, bw2[nt], c);
    const int col = nt * 16 + el;
#pragma unroll
    for (int r = 0; r < 8; ++r) {
      const int m  = r + (hi ? 8 : 0);
      const int dm = eidx[N_EDGES + e0 + m];
      atomicAdd(agg + (long)dm * 32 + col, c[r]);
    }
  }
}

// ---------------------------------------------------------------------------
// out += sum_n  W2 relu(W1 agg[n])   (block LDS reduction + global atomics)
// ---------------------------------------------------------------------------
__global__ void __launch_bounds__(256) final_kernel(
    const float* __restrict__ agg, const float* __restrict__ w1,
    const float* __restrict__ w2, float* __restrict__ out) {
  __shared__ float red[64];
  const int tid = threadIdx.x;
  if (tid < 64) red[tid] = 0.0f;
  __syncthreads();
  const long n = (long)blockIdx.x * blockDim.x + tid;
  if (n < N_NODES) {
    float a[32];
#pragma unroll
    for (int k = 0; k < 32; ++k) a[k] = agg[n * 32 + k];
    float hid[32];
#pragma unroll
    for (int o = 0; o < 32; ++o) {
      float s = 0.0f;
#pragma unroll
      for (int k = 0; k < 32; ++k) s = fmaf(a[k], w1[o * 32 + k], s);
      hid[o] = fmaxf(s, 0.0f);
    }
    for (int o = 0; o < 64; ++o) {
      float s = 0.0f;
#pragma unroll
      for (int k = 0; k < 32; ++k) s = fmaf(hid[k], w2[o * 32 + k], s);
      atomicAdd(&red[o], s);
    }
  }
  __syncthreads();
  if (tid < 64) atomicAdd(&out[tid], red[tid]);
}

// ---------------------------------------------------------------------------
extern "C" void kernel_launch(void* const* d_in, const int* in_sizes, int n_in,
                              void* d_out, int out_size, void* d_ws, size_t ws_size,
                              hipStream_t stream) {
  const float* x       = (const float*)d_in[0];
  const float* traj    = (const float*)d_in[1];
  const float* wih_f   = (const float*)d_in[2];
  const float* whh_f   = (const float*)d_in[3];
  const float* bih_f   = (const float*)d_in[4];
  const float* bhh_f   = (const float*)d_in[5];
  const float* wih_b   = (const float*)d_in[6];
  const float* whh_b   = (const float*)d_in[7];
  const float* bih_b   = (const float*)d_in[8];
  const float* bhh_b   = (const float*)d_in[9];
  const float* coor_w1 = (const float*)d_in[10];
  const float* coor_w2 = (const float*)d_in[11];
  const float* nn2_w1  = (const float*)d_in[12];
  const float* nn2_w2  = (const float*)d_in[13];
  const float* nn_w1   = (const float*)d_in[14];
  const float* nn_w2   = (const float*)d_in[15];
  const int*   eidx    = (const int*)d_in[16];
  float* out = (float*)d_out;

  char* ws = (char*)d_ws;
  size_t off = 0;
  auto alloc = [&](size_t bytes) {
    void* p = ws + off;
    off = (off + bytes + 255) & ~(size_t)255;
    return p;
  };
  _Float16* traj16 = (_Float16*)alloc((size_t)N_EDGES * 32 * sizeof(_Float16)); // 51.2 MB
  float*    agg    = (float*)   alloc((size_t)N_NODES * 32 * sizeof(float));    // 12.8 MB
  _Float16* coor16 = (_Float16*)alloc((size_t)N_NODES * 32 * sizeof(_Float16)); //  6.4 MB
  _Float16* w1h    = (_Float16*)alloc(32 * 96 * sizeof(_Float16));
  _Float16* w2h    = (_Float16*)alloc(32 * 32 * sizeof(_Float16));
  _Float16* whh16f = (_Float16*)alloc(128 * 32 * sizeof(_Float16));
  _Float16* whh16b = (_Float16*)alloc(128 * 32 * sizeof(_Float16));
  _Float16* wihpf  = (_Float16*)alloc(128 * 8 * sizeof(_Float16));
  _Float16* wihpb  = (_Float16*)alloc(128 * 8 * sizeof(_Float16));
  (void)in_sizes; (void)n_in; (void)out_size; (void)ws_size;

  init_kernel<<<4096, 256, 0, stream>>>(agg, out, nn2_w1, nn2_w2, w1h, w2h,
                                        whh_f, whh_b, wih_f, wih_b,
                                        bih_f, bhh_f, bih_b, bhh_b,
                                        whh16f, whh16b, wihpf, wihpb);
  coor_kernel<<<(N_NODES + 255) / 256, 256, 0, stream>>>(x, coor_w1, coor_w2, coor16);
  lstm_kernel<<<N_EDGES / 128, 256, 0, stream>>>(traj, whh16f, whh16b, wihpf, wihpb,
                                                 traj16);
  msg_kernel<<<N_EDGES / 128, 256, 0, stream>>>(eidx, coor16, traj16, w1h, w2h, agg);
  final_kernel<<<(N_NODES + 255) / 256, 256, 0, stream>>>(agg, nn_w1, nn_w2, out);
}
